// LanguageModel_89790586290626
// MI455X (gfx1250) — matrix operand order repaired and checked by reading, not compiled, not run on hardware
//
#include <hip/hip_runtime.h>
#include <hip/hip_bf16.h>
#include <stdint.h>

// Problem constants (from reference)
#define B_  64
#define T_  512
#define H_  300
#define V_  50257
#define KP  320            // K padded to multiple of 32 (WMMA bf16 K=32)
#define GP  304            // G/N padded to multiple of 16
#define NT_H  19           // GP/16
#define MT_BT 2048         // (B*T)/16
#define VT  3142           // ceil(V/16)

typedef __bf16 bf16;
typedef __attribute__((ext_vector_type(16))) __bf16 v16bf;
typedef __attribute__((ext_vector_type(8)))  __bf16 v8bf;
typedef __attribute__((ext_vector_type(8)))  float  v8f;

// Native f32->bf16 convert: clang lowers fptrunc to the hardware cvt on gfx1250
// (CDNA5 has full bf16 VALU support) -- avoids the 3-op software RNE sequence.
__device__ __forceinline__ bf16 f2bf(float x) { return (bf16)x; }

// ---------------- Kernel 0: pad+convert W,U -> bf16 [GP][KP], zero-filled ----
__global__ __launch_bounds__(256)
void prep_kernel(const float* __restrict__ W, const float* __restrict__ U,
                 bf16* __restrict__ Wb, bf16* __restrict__ Ub) {
  int i = blockIdx.x * 256 + threadIdx.x;
  if (i >= GP * KP) return;
  int g = i / KP, k = i % KP;
  bool in = (g < H_) & (k < H_);
  Wb[i] = f2bf(in ? W[g * H_ + k] : 0.f);
  Ub[i] = f2bf(in ? U[g * H_ + k] : 0.f);
}

// ---------------- Kernel 1: ux[t][b][g] = sum_h emb[tok[b,t],h] * U[g,h] -----
// One wave per 16x16 output tile. A-fragments gathered straight from emb (f32,
// converted in registers); B-fragments are contiguous 32B bf16 loads from Ub.
__global__ __launch_bounds__(256)
void ux_kernel(const int* __restrict__ tokens, const float* __restrict__ emb,
               const bf16* __restrict__ Ub, float* __restrict__ uxT) {
  int lane  = threadIdx.x & 31;
  int wave  = threadIdx.x >> 5;
  int tile  = blockIdx.x * 8 + wave;        // exact grid: MT_BT*NT_H tiles
  int mt = tile / NT_H, nt = tile % NT_H;
  int r0 = mt * 16, g0 = nt * 16;
  int nlane = lane & 15;
  int hi    = (lane >= 16);
  int koffA = hi ? 16 : 0;
  int koffB = hi ? 16 : 0;
  int tok   = tokens[r0 + nlane];
  const float* arow = emb + (size_t)tok * H_;
  int gcol = g0 + nlane;
  const bf16* brow = Ub + (size_t)gcol * KP + koffB;
  v8f acc = {};
  for (int kt = 0; kt < 10; ++kt) {
    int k0 = kt * 32;
    v16bf a, b;
    if (k0 + 32 <= H_) {                       // kt 0..8: branch-free fast path
#pragma unroll
      for (int e = 0; e < 8; ++e) {
        a[e]     = f2bf(arow[k0 + koffA + e]);
        a[e + 8] = f2bf(arow[k0 + koffA + 8 + e]);
      }
    } else {                                   // kt 9: K tail 288..319 guarded
#pragma unroll
      for (int e = 0; e < 16; ++e) {
        int k = k0 + koffA + e;
        a[e] = (k < H_) ? f2bf(arow[k]) : f2bf(0.f);
      }
    }
    const v8bf* bp = (const v8bf*)(brow + k0); // Ub zero-padded: no guards
    v8bf bl = bp[0], bh = bp[1];
#pragma unroll
    for (int e = 0; e < 8; ++e) { b[e] = bl[e]; b[e + 8] = bh[e]; }
    acc = __builtin_amdgcn_wmma_f32_16x16x32_bf16(false, a, false, b,
                                                  (short)0, acc, false, false);
  }
  if (gcol < H_) {
#pragma unroll
    for (int r = 0; r < 8; ++r) {
      int row = r0 + r + (hi ? 8 : 0);         // flat (b*T + t)
      int t = row & (T_ - 1);
      int bb = row >> 9;                       // /T_ (512)
      uxT[((size_t)t * B_ + bb) * H_ + gcol] = acc[r];
    }
  }
}

// ---------------- Kernel 2: persistent scan, 1 workgroup, h-state in LDS -----
// h[64][KP] bf16 in LDS (40KB). W streamed from L2 as bf16 B-fragments.
// Per step: 4x19=76 tiles over 32 waves; compute->sync->write->sync.
__global__ __launch_bounds__(1024)
void scan_kernel(const float* __restrict__ ihs, const bf16* __restrict__ Wb,
                 const float* __restrict__ uxT, bf16* __restrict__ hbF) {
  __shared__ bf16 hb[B_ * KP];                 // 40 KB
  int tid = threadIdx.x;
  int lane = tid & 31, wave = tid >> 5;
  for (int i = tid; i < B_ * KP; i += 1024) {
    int bb = i / KP, hh = i % KP;
    hb[i] = f2bf(hh < H_ ? ihs[bb * H_ + hh] : 0.f);
  }
  __syncthreads();
  int nlane = lane & 15;
  int hi    = (lane >= 16);
  int koffA = hi ? 16 : 0;
  int koffB = hi ? 16 : 0;
  for (int t = 0; t < T_; ++t) {
    float out[3][8];
#pragma unroll
    for (int q = 0; q < 3; ++q) {
      int tn = wave + q * 32;                  // wave-uniform guard: EXEC all-1s
      if (tn < 76) {
        int mt = tn / NT_H, nt = tn % NT_H;
        int b0 = mt * 16, g0 = nt * 16;
        int gcol = g0 + nlane;
        const bf16* arow = hb + (size_t)(b0 + nlane) * KP + koffA;
        const bf16* brow = Wb + (size_t)gcol * KP + koffB;
        v8f acc = {};
#pragma unroll
        for (int kt = 0; kt < 10; ++kt) {
          int k0 = kt * 32;
          v16bf a, b;
          const v8bf* ap = (const v8bf*)(arow + k0);
          const v8bf* bp = (const v8bf*)(brow + k0);
          v8bf al = ap[0], ah = ap[1];         // +0 and +8 bf16 = 2 x v8bf
          v8bf bl = bp[0], bh = bp[1];
#pragma unroll
          for (int e = 0; e < 8; ++e) {
            a[e] = al[e]; a[e + 8] = ah[e];
            b[e] = bl[e]; b[e + 8] = bh[e];
          }
          acc = __builtin_amdgcn_wmma_f32_16x16x32_bf16(false, a, false, b,
                                                        (short)0, acc, false, false);
        }
#pragma unroll
        for (int r = 0; r < 8; ++r) {
          int m = b0 + r + (hi ? 8 : 0);
          float ux = (gcol < H_) ? uxT[((size_t)t * B_ + m) * H_ + gcol] : 0.f;
          out[q][r] = tanhf(acc[r] + ux);      // pad cols: tanh(0+0)=0 -> stays 0
        }
      }
    }
    __syncthreads();                            // all reads of hb done
#pragma unroll
    for (int q = 0; q < 3; ++q) {
      int tn = wave + q * 32;
      if (tn < 76) {
        int mt = tn / NT_H, nt = tn % NT_H;
        int b0 = mt * 16, g0 = nt * 16;
        int gcol = g0 + nlane;
#pragma unroll
        for (int r = 0; r < 8; ++r) {
          int m = b0 + r + (hi ? 8 : 0);
          hb[(size_t)m * KP + gcol] = f2bf(out[q][r]);
        }
      }
    }
    __syncthreads();                            // hb updated for next step
  }
  for (int i = tid; i < B_ * KP; i += 1024) hbF[i] = hb[i];
}

// ---------------- Kernel 3: logits = h_last @ Wout^T + bout ------------------
// HBM-critical: streams 60MB Wout exactly once. One wave per 16-wide vocab
// tile, computing ALL 4 m-tiles, so each Wout element is loaded+converted once
// and reused across 4 WMMAs (acc = 4 x v8f = 32 VGPRs).
__global__ __launch_bounds__(256)
void out_kernel(const bf16* __restrict__ hbF, const float* __restrict__ Wout,
                const float* __restrict__ bout, float* __restrict__ logits) {
  int lane = threadIdx.x & 31;
  int wave = threadIdx.x >> 5;
  int nt = blockIdx.x * 8 + wave;              // wave-uniform tile id
  if (nt >= VT) return;                        // wave-uniform: EXEC stays all-1s
  int nlane = lane & 15;
  int hi    = (lane >= 16);
  int koffA = hi ? 16 : 0;
  int koffB = hi ? 16 : 0;
  int vcol = nt * 16 + nlane;
  bool vok = vcol < V_;
  const float* brow = Wout + (size_t)(vok ? vcol : 0) * H_;
  const bf16*  abase = hbF + (size_t)nlane * KP + koffA;
  v8f acc[4] = {};
  for (int kt = 0; kt < 10; ++kt) {
    int k0 = kt * 32;
    v16bf b;
    if (k0 + koffB + 16 <= H_) {               // contiguous 64B f32 load + cvt
#pragma unroll
      for (int e = 0; e < 16; ++e) b[e] = f2bf(brow[k0 + koffB + e]);
    } else {                                   // K tail guarded
#pragma unroll
      for (int e = 0; e < 16; ++e) {
        int k = k0 + koffB + e;
        b[e] = (k < H_) ? f2bf(brow[k]) : f2bf(0.f);
      }
    }
#pragma unroll
    for (int m4 = 0; m4 < 4; ++m4) {           // reuse B across 4 m-tiles
      const v8bf* ap = (const v8bf*)(abase + (size_t)(m4 * 16) * KP + k0);
      v8bf al = ap[0], ah = ap[1];
      v16bf a;
#pragma unroll
      for (int e = 0; e < 8; ++e) { a[e] = al[e]; a[e + 8] = ah[e]; }
      acc[m4] = __builtin_amdgcn_wmma_f32_16x16x32_bf16(false, a, false, b,
                                                        (short)0, acc[m4],
                                                        false, false);
    }
  }
  if (vok) {
    float bias = bout[vcol];
#pragma unroll
    for (int m4 = 0; m4 < 4; ++m4) {
#pragma unroll
      for (int r = 0; r < 8; ++r) {
        int m = m4 * 16 + r + (hi ? 8 : 0);
        logits[(size_t)m * V_ + vcol] = acc[m4][r] + bias;
      }
    }
  }
}

// ---------------- Host launcher ---------------------------------------------
extern "C" void kernel_launch(void* const* d_in, const int* in_sizes, int n_in,
                              void* d_out, int out_size, void* d_ws, size_t ws_size,
                              hipStream_t stream) {
  const float* ihs    = (const float*)d_in[0];
  const int*   tokens = (const int*)  d_in[1];
  const float* emb    = (const float*)d_in[2];
  const float* W      = (const float*)d_in[3];
  const float* U      = (const float*)d_in[4];
  const float* Wout   = (const float*)d_in[5];
  const float* bout   = (const float*)d_in[6];
  float* logits = (float*)d_out;

  // Workspace layout (bytes): Wb | Ub | uxT | hbF   (~37.9 MiB total)
  char* ws = (char*)d_ws;
  size_t oWb  = 0;
  size_t oUb  = oWb + (size_t)GP * KP * sizeof(bf16);       // 194560
  size_t oUx  = oUb + (size_t)GP * KP * sizeof(bf16);       // 389120
  size_t oHbF = oUx + (size_t)T_ * B_ * H_ * sizeof(float); // +39321600
  bf16*  Wb  = (bf16*)(ws + oWb);
  bf16*  Ub  = (bf16*)(ws + oUb);
  float* uxT = (float*)(ws + oUx);
  bf16*  hbF = (bf16*)(ws + oHbF);

  prep_kernel<<<(GP * KP + 255) / 256, 256, 0, stream>>>(W, U, Wb, Ub);
  ux_kernel<<<(MT_BT * NT_H) / 8, 256, 0, stream>>>(tokens, emb, Ub, uxT);
  scan_kernel<<<1, 1024, 0, stream>>>(ihs, Wb, uxT, hbF);
  out_kernel<<<(VT + 7) / 8, 256, 0, stream>>>(hbF, Wout, bout, logits);
}
